// QNet_86749749444727
// MI455X (gfx1250) — compile-verified, matrix operand order checked by reading, not run on hardware
//
#include <hip/hip_runtime.h>
#include <hip/hip_bf16.h>

// ---------------- problem constants ----------------
#define BATCH   4096
#define KNN_    8
#define NNODE   80          // KNN * (S + D)
#define NDC_    12
#define XSTRIDE 1376        // per-sample x width
#define KROW    172         // per-knn chunk in x
#define KFEAT   784         // feature width = KNN * 98
#define KPAD    800         // padded K for WMMA (25 steps of 32)
#define NHID    512
#define NACT    5

typedef __attribute__((ext_vector_type(16))) _Float16 half16_t;
typedef __attribute__((ext_vector_type(8)))  _Float16 half8_t;
typedef __attribute__((ext_vector_type(8)))  float    float8_t;

// ---------------- prep: W1 -> f16, K-padded ----------------
__global__ __launch_bounds__(128)
void prep_w1(const float* __restrict__ W1, _Float16* __restrict__ W1h) {
    const int n = blockIdx.x;                       // 0..511
    for (int k = threadIdx.x; k < KPAD; k += 128) {
        float v = (k < KFEAT) ? W1[(size_t)n * KFEAT + k] : 0.0f;
        W1h[(size_t)n * KPAD + k] = (_Float16)v;
    }
}

// ---------------- prep: noisy-net head weights ----------------
__device__ __forceinline__ float fsig(float t) {
    return copysignf(sqrtf(fabsf(t)), t);
}

// HW layout (floats): [0,2560) = w_adv (5x512), [2560,3072) = w_v (512),
//                     [3072,3077) = b_adv, [3077] = b_v
__global__ __launch_bounds__(256)
void prep_heads(const float* __restrict__ adv_uw, const float* __restrict__ adv_sw,
                const float* __restrict__ adv_ub, const float* __restrict__ adv_sb,
                const float* __restrict__ v_uw,   const float* __restrict__ v_sw,
                const float* __restrict__ v_ub,   const float* __restrict__ v_sb,
                const float* __restrict__ e_in_a, const float* __restrict__ e_out_a,
                const float* __restrict__ e_in_v, const float* __restrict__ e_out_v,
                float* __restrict__ HW) {
    const int tid = threadIdx.x;
    for (int j = tid; j < NHID; j += 256) {
        const float fa = fsig(e_in_a[j]);
        const float fv = fsig(e_in_v[j]);
        #pragma unroll
        for (int a = 0; a < NACT; ++a)
            HW[a * NHID + j] = adv_uw[a * NHID + j] + adv_sw[a * NHID + j] * fsig(e_out_a[a]) * fa;
        HW[NACT * NHID + j] = v_uw[j] + v_sw[j] * fsig(e_out_v[0]) * fv;
    }
    if (tid == 0) {
        #pragma unroll
        for (int a = 0; a < NACT; ++a)
            HW[3072 + a] = adv_ub[a] + adv_sb[a] * fsig(e_out_a[a]);
        HW[3077] = v_ub[0] + v_sb[0] * fsig(e_out_v[0]);
    }
}

// ---------------- per-sample GCN + feature assembly ----------------
// Sparse-adjacency insight: A[s,t] = delta(s,t) + cnt[s]*ADJ(nd[s],nd[t]) when
// both s,t are first occurrences; each column has <=5 nonzeros (self + 4 grid
// neighbors), so Anorm^T @ AG is ~5x80x12 MACs instead of 80x80x12.
__global__ __launch_bounds__(128)
void build_feat(const float* __restrict__ x, const float* __restrict__ Wg,
                const float* __restrict__ bg, _Float16* __restrict__ featA) {
    __shared__ float xs[XSTRIDE];
    __shared__ float wgs[144];
    __shared__ float bgs[12];
    __shared__ int   nds[NNODE];
    __shared__ int   firsts[NNODE];
    __shared__ float cnts[NNODE];
    __shared__ float diss[NNODE];
    __shared__ float AG[NNODE][NDC_];
    __shared__ float Hs[NNODE][NDC_];

    const int tid = threadIdx.x;
    const int b   = blockIdx.x;

    for (int i = tid; i < XSTRIDE; i += 128) xs[i] = x[(size_t)b * XSTRIDE + i];
    for (int i = tid; i < 144; i += 128)     wgs[i] = Wg[i];
    if (tid < 12) bgs[tid] = bg[tid];
    __syncthreads();

    // node ids + AG = all_info @ Wg
    if (tid < NNODE) {
        const int k = tid / 10, p = tid % 10, base = k * KROW;
        const float nv = (p < 5) ? xs[base + 160 + p] : xs[base + 165 + (p - 5)];
        nds[tid] = (int)nv;
        const float* ai = (p < 5) ? &xs[base + p * 16 + 4]
                                  : &xs[base + 80 + (p - 5) * 16 + 4];
        #pragma unroll
        for (int c = 0; c < NDC_; ++c) {
            float a = 0.f;
            #pragma unroll
            for (int d = 0; d < NDC_; ++d) a += ai[d] * wgs[d * 12 + c];
            AG[tid][c] = a;
        }
    }
    __syncthreads();

    // first occurrence + multiplicity
    if (tid < NNODE) {
        const int me = nds[tid];
        int f = tid; float cnt = 0.f;
        for (int s = 0; s < NNODE; ++s)
            if (nds[s] == me) { cnt += 1.f; if (s < f) f = s; }
        firsts[tid] = f; cnts[tid] = cnt;
    }
    __syncthreads();

    // degree -> rsqrt
    if (tid < NNODE) {
        const int me = nds[tid];
        const int r = me >> 5, c = me & 31;
        const int up = ((r + 31) & 31) * 32 + c, dn = ((r + 1) & 31) * 32 + c;
        const int lf = (r << 5) + ((c + 31) & 31), rt = (r << 5) + ((c + 1) & 31);
        float deg = 1.f;
        if (firsts[tid] == tid) {
            for (int s = 0; s < NNODE; ++s) {
                if (firsts[s] == s) {
                    const int v = nds[s];
                    if (v == up || v == dn || v == lf || v == rt) deg += cnts[s];
                }
            }
        }
        diss[tid] = rsqrtf(deg);
    }
    __syncthreads();

    // H = relu(Anorm^T @ AG + bg), sparse column accumulation
    if (tid < NNODE) {
        const int me = nds[tid];
        const int r = me >> 5, c2 = me & 31;
        const int up = ((r + 31) & 31) * 32 + c2, dn = ((r + 1) & 31) * 32 + c2;
        const int lf = (r << 5) + ((c2 + 31) & 31), rt = (r << 5) + ((c2 + 1) & 31);
        const float wj = diss[tid];
        float acc[NDC_];
        #pragma unroll
        for (int c = 0; c < NDC_; ++c) acc[c] = AG[tid][c] * wj * wj;
        if (firsts[tid] == tid) {
            for (int s = 0; s < NNODE; ++s) {
                if (s != tid && firsts[s] == s) {
                    const int v = nds[s];
                    if (v == up || v == dn || v == lf || v == rt) {
                        const float w = cnts[s] * diss[s] * wj;
                        #pragma unroll
                        for (int c = 0; c < NDC_; ++c) acc[c] += w * AG[s][c];
                    }
                }
            }
        }
        #pragma unroll
        for (int c = 0; c < NDC_; ++c) {
            const float h = acc[c] + bgs[c];
            Hs[tid][c] = h > 0.f ? h : 0.f;
        }
    }
    __syncthreads();

    // gather -> 784-wide feature row (f16, zero-padded to 800)
    _Float16* orow = featA + (size_t)b * KPAD;
    for (int col = tid; col < KPAD; col += 128) {
        float val = 0.f;
        if (col < KFEAT) {
            const int k = col / 98, rr = col % 98, base = k * KROW;
            if (rr < 72) {                       // gnn_feat: H[first[k*10+a], c]
                const int a = rr / 12, c = rr % 12;
                val = Hs[firsts[k * 10 + a]][c];
            } else if (rr < 74) {                // els
                val = xs[base + 170 + (rr - 72)];
            } else {                             // each_feat: local first match
                const int e4 = rr - 74, a = e4 >> 2, e = e4 & 3;
                const int tgt = nds[k * 10 + a];
                int loc = 0;
                for (int j = 0; j < 10; ++j)
                    if (nds[k * 10 + j] == tgt) { loc = j; break; }
                val = (loc < 5) ? xs[base + loc * 16 + e]
                                : xs[base + 80 + (loc - 5) * 16 + e];
            }
        }
        orow[col] = (_Float16)val;
    }
}

// ---------------- WMMA GEMM (h = relu(feat@W1^T+b1)) + fused heads ----------------
// Block: 256 threads (8 waves). Tile: 16 batch rows x 512 cols.
// Wave w owns N-chunk [w*64, w*64+64) = 4 WMMA tiles; K loop 25 x 32.
__global__ __launch_bounds__(256)
void gemm_head(const _Float16* __restrict__ A, const _Float16* __restrict__ Bw,
               const float* __restrict__ b1, const float* __restrict__ HW,
               float* __restrict__ out) {
    __shared__ float hsh[16 * NHID];               // 32 KB

    const int tid  = threadIdx.x;
    const int lane = tid & 31;
    const int wave = tid >> 5;
    const int m0   = blockIdx.x * 16;
    const int nbase = wave * 64;
    const int half  = lane & 15;                   // M-row (A) / N-col (B) in tile
    const int khalf = (lane >> 4) * 8;             // K sub-group per 16-bit layout

    float8_t acc[4] = {};

    const _Float16* arow = A + (size_t)(m0 + half) * KPAD + khalf;

    for (int k0 = 0; k0 < KPAD; k0 += 32) {
        __builtin_prefetch(arow + k0 + 128, 0, 1);
        const half8_t alo = *(const half8_t*)(arow + k0);
        const half8_t ahi = *(const half8_t*)(arow + k0 + 16);
        half16_t af;
        #pragma unroll
        for (int i = 0; i < 8; ++i) { af[i] = alo[i]; af[i + 8] = ahi[i]; }

        #pragma unroll
        for (int t = 0; t < 4; ++t) {
            const _Float16* brow = Bw + (size_t)(nbase + t * 16 + half) * KPAD + khalf + k0;
            __builtin_prefetch(brow + 128, 0, 1);
            const half8_t blo = *(const half8_t*)(brow);
            const half8_t bhi = *(const half8_t*)(brow + 16);
            half16_t bf;
            #pragma unroll
            for (int i = 0; i < 8; ++i) { bf[i] = blo[i]; bf[i + 8] = bhi[i]; }
            acc[t] = __builtin_amdgcn_wmma_f32_16x16x32_f16(
                false, af, false, bf, (short)0, acc[t], false, false);
        }
    }

    // epilogue: +b1, relu, stash h tile in LDS (C layout: N=lane&15, M=(lane>>4)*8+r)
    #pragma unroll
    for (int t = 0; t < 4; ++t) {
        const int col = nbase + t * 16 + half;
        const float bb = b1[col];
        #pragma unroll
        for (int r = 0; r < 8; ++r) {
            const int row = (lane >> 4) * 8 + r;
            const float v = acc[t][r] + bb;
            hsh[row * NHID + col] = v > 0.f ? v : 0.f;
        }
    }
    __syncthreads();

    // fused noisy dueling heads: 6 dots of 512 per row, wave32 shfl reduction
    #pragma unroll
    for (int rr = 0; rr < 2; ++rr) {
        const int row = wave * 2 + rr;
        float a[6] = {0.f, 0.f, 0.f, 0.f, 0.f, 0.f};
        for (int c = lane; c < NHID; c += 32) {
            const float hv = hsh[row * NHID + c];
            #pragma unroll
            for (int j = 0; j < 6; ++j) a[j] += hv * HW[j * NHID + c];
        }
        #pragma unroll
        for (int off = 16; off > 0; off >>= 1) {
            #pragma unroll
            for (int j = 0; j < 6; ++j) a[j] += __shfl_xor(a[j], off, 32);
        }
        if (lane == 0) {
            float adv[NACT], m = 0.f;
            #pragma unroll
            for (int j = 0; j < NACT; ++j) { adv[j] = a[j] + HW[3072 + j]; m += adv[j]; }
            m *= 0.2f;
            const float v = a[5] + HW[3077];
            float* o = out + (size_t)(m0 + row) * NACT;
            #pragma unroll
            for (int j = 0; j < NACT; ++j) o[j] = v + adv[j] - m;
        }
    }
}

// ---------------- launch ----------------
extern "C" void kernel_launch(void* const* d_in, const int* in_sizes, int n_in,
                              void* d_out, int out_size, void* d_ws, size_t ws_size,
                              hipStream_t stream) {
    const float* x      = (const float*)d_in[0];
    const float* Wg     = (const float*)d_in[1];
    const float* bg     = (const float*)d_in[2];
    const float* W1     = (const float*)d_in[3];
    const float* b1     = (const float*)d_in[4];
    const float* adv_uw = (const float*)d_in[5];
    const float* adv_sw = (const float*)d_in[6];
    const float* adv_ub = (const float*)d_in[7];
    const float* adv_sb = (const float*)d_in[8];
    const float* v_uw   = (const float*)d_in[9];
    const float* v_sw   = (const float*)d_in[10];
    const float* v_ub   = (const float*)d_in[11];
    const float* v_sb   = (const float*)d_in[12];
    const float* e_in_a = (const float*)d_in[13];
    const float* e_out_a= (const float*)d_in[14];
    const float* e_in_v = (const float*)d_in[15];
    const float* e_out_v= (const float*)d_in[16];

    char* ws = (char*)d_ws;
    _Float16* featA = (_Float16*)ws;                                   // 4096*800*2 B
    _Float16* W1h   = (_Float16*)(ws + (size_t)BATCH * KPAD * 2);      // 512*800*2 B
    float*    HW    = (float*)(ws + (size_t)BATCH * KPAD * 2
                                  + (size_t)NHID * KPAD * 2);          // ~12.3 KB
    float* out = (float*)d_out;

    prep_w1   <<<NHID, 128, 0, stream>>>(W1, W1h);
    prep_heads<<<1, 256, 0, stream>>>(adv_uw, adv_sw, adv_ub, adv_sb,
                                      v_uw, v_sw, v_ub, v_sb,
                                      e_in_a, e_out_a, e_in_v, e_out_v, HW);
    build_feat<<<BATCH, 128, 0, stream>>>(x, Wg, bg, featA);
    gemm_head <<<BATCH / 16, 256, 0, stream>>>(featA, W1h, b1, HW, out);
}
